// AOI_20151986552919
// MI455X (gfx1250) — compile-verified, matrix operand order checked by reading, not run on hardware
//
#include <hip/hip_runtime.h>
#include <hip/hip_bf16.h>
#include <math.h>

// ---------------------------------------------------------------------------
// AOI option-compare network on MI455X (gfx1250).
// All GEMMs run on v_wmma_f32_16x16x32_bf16 (wave32), fp32 accumulate.
// Wave tile 32x64 (8 WMMAs per k-step, each LDS B-fragment reused twice).
// ---------------------------------------------------------------------------

#define Hd    768
#define Sdim  256
#define Bopt  128
#define CHB   32          // options per pipeline chunk (must be multiple of 4)
#define NCHK  (Bopt / CHB)

typedef __attribute__((ext_vector_type(16))) __bf16          v16bf;
typedef __attribute__((ext_vector_type(16))) unsigned short  v16u;
typedef __attribute__((ext_vector_type(8)))  float           v8f;

__device__ __forceinline__ unsigned short f2bf(float f) {
  union { float f; unsigned u; } v; v.f = f;
  unsigned r = v.u + 0x7fffu + ((v.u >> 16) & 1u);   // round-to-nearest-even
  return (unsigned short)(r >> 16);
}
__device__ __forceinline__ float bf2f(unsigned short h) {
  union { unsigned u; float f; } v; v.u = ((unsigned)h) << 16;
  return v.f;
}
// sibling option index: others_tab[b%4][k] within the 4-option group
__device__ __forceinline__ int other_idx(int b, int k) {
  int r = b & 3;
  int o = k + (k >= r ? 1 : 0);
  return (b & ~3) + o;
}

// 16x32 bf16 fragment (A-matrix layout; also B for NT when source is [N,K]
// row-major, and B for NN when source was transposed into LDS as [n][k]).
// Lane l: m = l&15, h = l>>4.  elem e=2j+bit:
//   j<4 : K = 8h + 2j + bit      j>=4 : K = 8h + 16 + 2(j-4) + bit
__device__ __forceinline__ v16bf frag_rm(const unsigned short* p, int ld) {
  int lane = threadIdx.x & 31;
  const unsigned short* r = p + (size_t)(lane & 15) * ld + ((lane >> 4) << 3);
  v16u t;
#pragma unroll
  for (int j = 0; j < 4; ++j) {
    t[2 * j]     = r[2 * j];
    t[2 * j + 1] = r[2 * j + 1];
    t[8 + 2 * j]     = r[16 + 2 * j];
    t[8 + 2 * j + 1] = r[16 + 2 * j + 1];
  }
  return __builtin_bit_cast(v16bf, t);
}

__device__ __forceinline__ v8f wmma_bf16(v16bf a, v16bf b, v8f c) {
  return __builtin_amdgcn_wmma_f32_16x16x32_bf16(
      /*neg_a=*/false, a, /*neg_b=*/false, b,
      /*c_mod=*/(short)0, c, /*reuse_a=*/false, /*reuse_b=*/false);
}

// ---------------------------------------------------------------------------
__global__ void k_f32_to_bf16(const float* __restrict__ s,
                              unsigned short* __restrict__ d, int n) {
  for (int i = blockIdx.x * blockDim.x + threadIdx.x; i < n;
       i += gridDim.x * blockDim.x)
    d[i] = f2bf(s[i]);
}

// stage a 64n x 32k bf16 weight tile into LDS (row-major [n][32]),
// 128 threads, fixed trip count, b128 loads/stores.
__device__ __forceinline__ void stage_w64(const unsigned short* __restrict__ W,
                                          int ldw, int n0, int kk,
                                          unsigned short* __restrict__ sW) {
#pragma unroll
  for (int it = 0; it < 2; ++it) {
    int c = threadIdx.x + it * 128;          // 256 chunks of 16B
    int n = c >> 2, part = c & 3;
    const uint4* g =
        reinterpret_cast<const uint4*>(W + (size_t)(n0 + n) * ldw + kk) + part;
    *reinterpret_cast<uint4*>(&sW[n * 32 + part * 8]) = *g;
  }
}

// ---------------------------------------------------------------------------
// Generic NT GEMM:  C_bf16[M,N] = A_bf16[M,K] @ W_bf16[N,K]^T + bias (+relu)
// block = 128 threads (4 waves); block tile 128x64; wave tile 32x64.
__global__ void k_gemm_nt(const unsigned short* __restrict__ A, int lda,
                          const unsigned short* __restrict__ W, int ldw,
                          const float* __restrict__ bias,
                          unsigned short* __restrict__ C, int ldc,
                          int K, int relu) {
  __shared__ unsigned short sW[64 * 32];
  const int n0 = blockIdx.x * 64;
  const int m0 = blockIdx.y * 128;
  const int wave = threadIdx.x >> 5;
  v8f acc[2][4] = {};
  for (int kk = 0; kk < K; kk += 32) {
    stage_w64(W, ldw, n0, kk, sW);
    __syncthreads();
    const unsigned short* Ap = A + (size_t)(m0 + wave * 32) * lda + kk;
    v16bf a0 = frag_rm(Ap, lda);
    v16bf a1 = frag_rm(Ap + (size_t)16 * lda, lda);
#pragma unroll
    for (int i = 0; i < 4; ++i) {
      v16bf bf = frag_rm(&sW[(i * 16) * 32], 32);
      acc[0][i] = wmma_bf16(a0, bf, acc[0][i]);
      acc[1][i] = wmma_bf16(a1, bf, acc[1][i]);
    }
    __syncthreads();
  }
  int lane = threadIdx.x & 31, n = lane & 15, hh = lane >> 4;
#pragma unroll
  for (int r = 0; r < 2; ++r)
#pragma unroll
    for (int i = 0; i < 4; ++i) {
      int gn = n0 + i * 16 + n;
      float bv = bias ? bias[gn] : 0.f;
#pragma unroll
      for (int j = 0; j < 8; ++j) {
        int gm = m0 + wave * 32 + r * 16 + hh * 8 + j;
        float v = acc[r][i][j] + bv;
        if (relu) v = fmaxf(v, 0.f);
        C[(size_t)gm * ldc + gn] = f2bf(v);
      }
    }
}

// ---------------------------------------------------------------------------
// att[b,k] = p[b] @ trans_p[qb]^T  fused with the reference masked softmax.
// block = 128 threads; 32 rows x full 256 cols (wave w -> cols w*64..w*64+63).
// Emits sm bf16 [CHB*3, S, S].
__global__ void k_attn(const unsigned short* __restrict__ pb,
                       const unsigned short* __restrict__ tq,
                       const int* __restrict__ olen,
                       unsigned short* __restrict__ sm, int b_base) {
  __shared__ unsigned short sB[Sdim * 32];   // tq tile [256 t][32 k] (16 KB)
  __shared__ float sAtt[32 * Sdim];          // score rows (32 KB)
  __shared__ float sStat[32 * 2];            // per-row {max, denom}
  const int m0  = blockIdx.x * 32;
  const int lbk = blockIdx.y;
  const int b   = b_base + lbk / 3;
  const int qb  = other_idx(b, lbk % 3);
  const int wave = threadIdx.x >> 5;
  const unsigned short* Ab = pb + ((size_t)b * Sdim + m0) * Hd;
  const unsigned short* Qb = tq + (size_t)qb * Sdim * Hd;
  v8f acc[2][4] = {};
  for (int kk = 0; kk < Hd; kk += 32) {
#pragma unroll
    for (int it = 0; it < 8; ++it) {         // 1024 chunks of 16B
      int c = threadIdx.x + it * 128;
      int t = c >> 2, part = c & 3;
      const uint4* g =
          reinterpret_cast<const uint4*>(Qb + (size_t)t * Hd + kk) + part;
      *reinterpret_cast<uint4*>(&sB[t * 32 + part * 8]) = *g;
    }
    __syncthreads();
    v16bf a0 = frag_rm(Ab + kk, Hd);
    v16bf a1 = frag_rm(Ab + (size_t)16 * Hd + kk, Hd);
#pragma unroll
    for (int i = 0; i < 4; ++i) {
      v16bf bf = frag_rm(&sB[(wave * 64 + i * 16) * 32], 32);
      acc[0][i] = wmma_bf16(a0, bf, acc[0][i]);
      acc[1][i] = wmma_bf16(a1, bf, acc[1][i]);
    }
    __syncthreads();
  }
  {
    int lane = threadIdx.x & 31, n = lane & 15, hh = lane >> 4;
#pragma unroll
    for (int r = 0; r < 2; ++r)
#pragma unroll
      for (int i = 0; i < 4; ++i)
#pragma unroll
        for (int j = 0; j < 8; ++j)
          sAtt[(r * 16 + hh * 8 + j) * Sdim + wave * 64 + i * 16 + n] =
              acc[r][i][j];
  }
  __syncthreads();
  const int len = olen[qb] + 1;
  // reference: sm = softmax(att*mask)*mask, renorm by (sum + 1e-13)
  //   == exp(y-M)*mask / (S_mask + Z_full*1e-13),  y = att*mask, M = max(y)
  if (threadIdx.x < 32) {
    const float* row = &sAtt[threadIdx.x * Sdim];
    float mx = 0.f;                              // masked scores are exactly 0
    for (int t = 0; t < len; ++t) mx = fmaxf(mx, row[t]);
    float zfull = 0.f, zmask = 0.f;
    for (int t = 0; t < Sdim; ++t) {
      float y = (t < len) ? row[t] : 0.f;
      float e = __expf(y - mx);
      zfull += e;
      if (t < len) zmask += e;
    }
    sStat[threadIdx.x * 2]     = mx;
    sStat[threadIdx.x * 2 + 1] = zmask + zfull * 1e-13f;
  }
  __syncthreads();
  unsigned short* out = sm + ((size_t)lbk * Sdim + m0) * Sdim;
#pragma unroll
  for (int it = 0; it < 64; ++it) {
    int idx = threadIdx.x + it * 128;            // 8192 elements
    int r = idx >> 8, t = idx & 255;
    float v = 0.f;
    if (t < len)
      v = __expf(sAtt[r * Sdim + t] - sStat[r * 2]) / sStat[r * 2 + 1];
    out[idx] = f2bf(v);
  }
}

// ---------------------------------------------------------------------------
// av[b,k] = sm[b,k] @ q   (NN GEMM, K = S).  q tile transposed into LDS so
// B fragments become pair-contiguous ds reads.  Block tile 128x64.
__global__ void k_av(const unsigned short* __restrict__ sm,
                     const unsigned short* __restrict__ pb,
                     unsigned short* __restrict__ av, int b_base) {
  __shared__ unsigned short sB[64 * 32];     // [n][k]
  const int n0  = blockIdx.x * 64;           // over H
  const int m0  = blockIdx.y * 128;          // over S
  const int lbk = blockIdx.z;
  const int qb  = other_idx(b_base + lbk / 3, lbk % 3);
  const int wave = threadIdx.x >> 5;
  const unsigned short* A = sm + (size_t)lbk * Sdim * Sdim;
  const unsigned short* Q = pb + (size_t)qb * Sdim * Hd;   // [K=S, N=H]
  v8f acc[2][4] = {};
  for (int kk = 0; kk < Sdim; kk += 32) {
#pragma unroll
    for (int it = 0; it < 2; ++it) {         // 32 k-rows x 8 chunks of 8
      int c = threadIdx.x + it * 128;
      int kr = c >> 3, part = c & 7;
      const unsigned short* g = Q + (size_t)(kk + kr) * Hd + n0 + part * 8;
#pragma unroll
      for (int e = 0; e < 8; ++e) sB[(part * 8 + e) * 32 + kr] = g[e];
    }
    __syncthreads();
    const unsigned short* Ap = A + (size_t)(m0 + wave * 32) * Sdim + kk;
    v16bf a0 = frag_rm(Ap, Sdim);
    v16bf a1 = frag_rm(Ap + (size_t)16 * Sdim, Sdim);
#pragma unroll
    for (int i = 0; i < 4; ++i) {
      v16bf bf = frag_rm(&sB[(i * 16) * 32], 32);
      acc[0][i] = wmma_bf16(a0, bf, acc[0][i]);
      acc[1][i] = wmma_bf16(a1, bf, acc[1][i]);
    }
    __syncthreads();
  }
  int lane = threadIdx.x & 31, n = lane & 15, hh = lane >> 4;
  unsigned short* C = av + (size_t)lbk * Sdim * Hd;
#pragma unroll
  for (int r = 0; r < 2; ++r)
#pragma unroll
    for (int i = 0; i < 4; ++i)
#pragma unroll
      for (int j = 0; j < 8; ++j)
        C[(size_t)(m0 + wave * 32 + r * 16 + hh * 8 + j) * Hd + n0 + i * 16 + n] =
            f2bf(acc[r][i][j]);
}

// ---------------------------------------------------------------------------
// oc = cat(oc1,oc2,oc3) @ Wd3^T + bd3.  A rows gathered from [lb,kblk,s,h]
// layout (K chunks of 32 never straddle an H boundary since H%32==0).
__global__ void k_oc(const unsigned short* __restrict__ ock,   // [CHB*3,S,H]
                     const unsigned short* __restrict__ Wd3b,  // [H,3H]
                     const float* __restrict__ bd3,
                     unsigned short* __restrict__ oc) {        // [CHB*S,H]
  __shared__ unsigned short sW[64 * 32];
  const int n0 = blockIdx.x * 64;
  const int m0 = blockIdx.y * 128;
  const int lb = m0 >> 8, s0 = m0 & 255;
  const int wave = threadIdx.x >> 5;
  v8f acc[2][4] = {};
  for (int kk = 0; kk < 3 * Hd; kk += 32) {
    stage_w64(Wd3b, 3 * Hd, n0, kk, sW);
    __syncthreads();
    int kblk = kk / Hd, h0 = kk % Hd;
    const unsigned short* Ap =
        ock + (((size_t)lb * 3 + kblk) * Sdim + s0 + wave * 32) * Hd + h0;
    v16bf a0 = frag_rm(Ap, Hd);
    v16bf a1 = frag_rm(Ap + (size_t)16 * Hd, Hd);
#pragma unroll
    for (int i = 0; i < 4; ++i) {
      v16bf bf = frag_rm(&sW[(i * 16) * 32], 32);
      acc[0][i] = wmma_bf16(a0, bf, acc[0][i]);
      acc[1][i] = wmma_bf16(a1, bf, acc[1][i]);
    }
    __syncthreads();
  }
  int lane = threadIdx.x & 31, n = lane & 15, hh = lane >> 4;
#pragma unroll
  for (int r = 0; r < 2; ++r)
#pragma unroll
    for (int i = 0; i < 4; ++i) {
      int gn = n0 + i * 16 + n;
      float bv = bd3[gn];
#pragma unroll
      for (int j = 0; j < 8; ++j)
        oc[(size_t)(m0 + wave * 32 + r * 16 + hh * 8 + j) * Hd + gn] =
            f2bf(acc[r][i][j] + bv);
    }
}

// ---------------------------------------------------------------------------
// Fuse net: z = oc@W1^T + p@W2^T + b1 + b2 (single K=1536 concat contraction),
// mid = sigmoid(z), out = p*mid + oc*(1-mid).  Writes f32 d_out.
__global__ void k_fuse(const unsigned short* __restrict__ occ,  // [CHB*S,H]
                       const unsigned short* __restrict__ pbc,  // [CHB*S,H]
                       const unsigned short* __restrict__ W1b,
                       const unsigned short* __restrict__ W2b,
                       const float* __restrict__ b1,
                       const float* __restrict__ b2,
                       const float* __restrict__ pin,
                       float* __restrict__ outp) {
  __shared__ unsigned short sW[64 * 32];
  const int n0 = blockIdx.x * 64;
  const int m0 = blockIdx.y * 128;
  const int wave = threadIdx.x >> 5;
  v8f acc[2][4] = {};
  for (int kk = 0; kk < 2 * Hd; kk += 32) {
    const unsigned short* Wsrc = (kk < Hd) ? W1b : W2b;
    const int kloc = (kk < Hd) ? kk : kk - Hd;
    stage_w64(Wsrc, Hd, n0, kloc, sW);
    __syncthreads();
    const unsigned short* Asrc = (kk < Hd) ? occ : pbc;
    const unsigned short* Ap = Asrc + (size_t)(m0 + wave * 32) * Hd + kloc;
    v16bf a0 = frag_rm(Ap, Hd);
    v16bf a1 = frag_rm(Ap + (size_t)16 * Hd, Hd);
#pragma unroll
    for (int i = 0; i < 4; ++i) {
      v16bf bf = frag_rm(&sW[(i * 16) * 32], 32);
      acc[0][i] = wmma_bf16(a0, bf, acc[0][i]);
      acc[1][i] = wmma_bf16(a1, bf, acc[1][i]);
    }
    __syncthreads();
  }
  int lane = threadIdx.x & 31, n = lane & 15, hh = lane >> 4;
#pragma unroll
  for (int r = 0; r < 2; ++r)
#pragma unroll
    for (int i = 0; i < 4; ++i) {
      int gn = n0 + i * 16 + n;
      float bv = b1[gn] + b2[gn];
#pragma unroll
      for (int j = 0; j < 8; ++j) {
        size_t gidx = (size_t)(m0 + wave * 32 + r * 16 + hh * 8 + j) * Hd + gn;
        float mid = 1.f / (1.f + __expf(-(acc[r][i][j] + bv)));
        float pv = pin[gidx];
        float ov = bf2f(occ[gidx]);
        outp[gidx] = pv * mid + ov * (1.f - mid);
      }
    }
}

// ---------------------------------------------------------------------------
extern "C" void kernel_launch(void* const* d_in, const int* in_sizes, int n_in,
                              void* d_out, int out_size, void* d_ws,
                              size_t ws_size, hipStream_t stream) {
  (void)in_sizes; (void)n_in; (void)out_size; (void)ws_size;
  const float* p    = (const float*)d_in[0];
  const int*   olen = (const int*)  d_in[1];
  const float* Wt   = (const float*)d_in[2];
  const float* bt   = (const float*)d_in[3];
  const float* Wd   = (const float*)d_in[4];
  const float* bd   = (const float*)d_in[5];
  const float* Wd3  = (const float*)d_in[6];
  const float* bd3  = (const float*)d_in[7];
  const float* W1   = (const float*)d_in[8];
  const float* b1   = (const float*)d_in[9];
  const float* W2   = (const float*)d_in[10];
  const float* b2   = (const float*)d_in[11];
  float* out = (float*)d_out;

  // workspace carve-up (all bf16 raw ushort), ~210 MB total
  char* w = (char*)d_ws;
  auto take = [&](size_t elems) {
    unsigned short* r = (unsigned short*)w;
    w += ((elems * 2 + 255) & ~(size_t)255);
    return r;
  };
  const size_t PSZ = (size_t)Bopt * Sdim * Hd;
  unsigned short* pb   = take(PSZ);                       // bf16(p)
  unsigned short* tq   = take(PSZ);                       // trans_p bf16
  unsigned short* Wtb  = take((size_t)Hd * Hd);
  unsigned short* Wdb  = take((size_t)Hd * Hd);
  unsigned short* W1b  = take((size_t)Hd * Hd);
  unsigned short* W2b  = take((size_t)Hd * Hd);
  unsigned short* Wd3b = take((size_t)Hd * 3 * Hd);
  unsigned short* smw  = take((size_t)CHB * 3 * Sdim * Sdim);
  unsigned short* avw  = take((size_t)CHB * 3 * Sdim * Hd);
  unsigned short* ockw = take((size_t)CHB * 3 * Sdim * Hd);
  unsigned short* occw = take((size_t)CHB * Sdim * Hd);

  // fp32 -> bf16 conversions
  k_f32_to_bf16<<<2048, 256, 0, stream>>>(p,   pb,   (int)PSZ);
  k_f32_to_bf16<<<256,  256, 0, stream>>>(Wt,  Wtb,  Hd * Hd);
  k_f32_to_bf16<<<256,  256, 0, stream>>>(Wd,  Wdb,  Hd * Hd);
  k_f32_to_bf16<<<256,  256, 0, stream>>>(W1,  W1b,  Hd * Hd);
  k_f32_to_bf16<<<256,  256, 0, stream>>>(W2,  W2b,  Hd * Hd);
  k_f32_to_bf16<<<512,  256, 0, stream>>>(Wd3, Wd3b, Hd * 3 * Hd);

  // trans_p = p @ Wt^T + bt   (shared by all 3 comparison slots)
  k_gemm_nt<<<dim3(Hd / 64, (Bopt * Sdim) / 128), 128, 0, stream>>>(
      pb, Hd, Wtb, Hd, bt, tq, Hd, Hd, 0);

  for (int cb = 0; cb < NCHK; ++cb) {
    const int b0 = cb * CHB;
    const size_t poff = (size_t)b0 * Sdim * Hd;
    // scores + masked softmax
    k_attn<<<dim3(Sdim / 32, CHB * 3), 128, 0, stream>>>(pb, tq, olen, smw, b0);
    // att_vec = sm @ q
    k_av<<<dim3(Hd / 64, Sdim / 128, CHB * 3), 128, 0, stream>>>(smw, pb, avw,
                                                                 b0);
    // oc_k = relu(av @ Wd^T + bd)
    k_gemm_nt<<<dim3(Hd / 64, (CHB * 3 * Sdim) / 128), 128, 0, stream>>>(
        avw, Hd, Wdb, Hd, bd, ockw, Hd, Hd, 1);
    // oc = cat @ Wd3^T + bd3
    k_oc<<<dim3(Hd / 64, (CHB * Sdim) / 128), 128, 0, stream>>>(ockw, Wd3b, bd3,
                                                                occw);
    // fuse net + output
    k_fuse<<<dim3(Hd / 64, (CHB * Sdim) / 128), 128, 0, stream>>>(
        occw, pb + poff, W1b, W2b, b1, b2, p + poff, out + poff);
  }
}
